// MultiHeadAttentionLayer_35759897707169
// MI455X (gfx1250) — matrix operand order repaired and checked by reading, not compiled, not run on hardware
//
#include <hip/hip_runtime.h>

// ---------------------------------------------------------------------------
// MHA for MI455X (gfx1250): bf16 WMMA GEMMs + flash attention, wave32.
// K-blocks stream into LDS via global_load_async_to_lds_b128 (ASYNCcnt path).
// ---------------------------------------------------------------------------

#define D_MODEL 1024
#define SEQ     2048
#define HEADS   16
#define DK      64
#define BATCH   2

typedef __attribute__((ext_vector_type(16))) __bf16 v16bf;
typedef __attribute__((ext_vector_type(8)))  float  v8f;
typedef unsigned short u16t;
typedef __attribute__((ext_vector_type(8)))  u16t   u16x8;

union BF16Frag {
  v16bf v;
  u16x8 h[2];
};

__device__ __forceinline__ u16t bf16_of(float x) {
  union { float f; unsigned u; } c; c.f = x;
  unsigned r = c.u + 0x7FFFu + ((c.u >> 16) & 1u);   // round-to-nearest-even
  return (u16t)(r >> 16);
}

__device__ __forceinline__ v8f wmma_bf16(v16bf a, v16bf b, v8f c) {
  // D = A(16x32 bf16) * B(32x16 bf16) + C(16x16 f32)
  return __builtin_amdgcn_wmma_f32_16x16x32_bf16(false, a, false, b, (short)0, c,
                                                 false, false);
}

__device__ __forceinline__ void wait_asynccnt0() {
#if __has_builtin(__builtin_amdgcn_s_wait_asynccnt)
  __builtin_amdgcn_s_wait_asynccnt(0);
#else
  asm volatile("s_wait_asynccnt 0x0" ::: "memory");
#endif
}

// ---------------------------------------------------------------------------
// C[M,N] = A[M,K] * W[K,N] + bias.  M=4096, K=N=1024.
// Tile: 128x128 per workgroup (256 thr = 8 waves, 4x2 wave grid, 32x64/wave),
// BK=32 (one WMMA k-step). A & W staged to LDS as bf16 (W transposed so B
// fragments are contiguous ds_load_b128s).
// SPLIT_BF16: write bf16 into [B,H,S,DK] split-head layout, else f32 [M,N].
// ---------------------------------------------------------------------------
template <bool SPLIT_BF16>
__global__ __launch_bounds__(256)
void gemm_bias(const float* __restrict__ A, const float* __restrict__ W,
               const float* __restrict__ bias, void* __restrict__ outp) {
  constexpr int K = D_MODEL, N = D_MODEL;
  constexpr int BM = 128, BK = 32;
  constexpr int LDA = BK + 8;   // u16 stride, 80B rows -> 16B aligned
  constexpr int LDW = BK + 8;
  __shared__ alignas(16) u16t As[BM * LDA];
  __shared__ alignas(16) u16t Wt[128 * LDW];   // Wt[n][k]

  const int tid  = threadIdx.x;
  const int lane = tid & 31;
  const int wid  = tid >> 5;
  const int wm   = wid >> 1;          // 0..3 -> 32-row strip
  const int wn   = wid & 1;           // 0..1 -> 64-col strip
  const int hl   = lane >> 4;         // lane half
  const int l15  = lane & 15;
  const int tile_m = blockIdx.x * BM;
  const int tile_n = blockIdx.y * 128;

  v8f acc[2][4] = {};

  const int a_row = tid >> 1;            // 0..127
  const int a_k   = (tid & 1) * 16;      // 0 or 16
  const int w_k   = tid >> 3;            // 0..31
  const int w_n   = (tid & 7) * 16;      // 0..112

  for (int k0 = 0; k0 < K; k0 += BK) {
    // ---- stage A tile (128 x 32) f32 -> bf16, row-major ----
    {
      const float4* src = reinterpret_cast<const float4*>(
          A + (size_t)(tile_m + a_row) * K + (k0 + a_k));
      u16x8 pk[2];
#pragma unroll
      for (int i = 0; i < 2; ++i) {
        float4 v0 = src[2 * i], v1 = src[2 * i + 1];
        pk[i] = u16x8{bf16_of(v0.x), bf16_of(v0.y), bf16_of(v0.z), bf16_of(v0.w),
                      bf16_of(v1.x), bf16_of(v1.y), bf16_of(v1.z), bf16_of(v1.w)};
      }
      u16x8* dst = reinterpret_cast<u16x8*>(&As[a_row * LDA + a_k]);
      dst[0] = pk[0];
      dst[1] = pk[1];
    }
    // ---- stage W tile (32 x 128) transposed into Wt[n][k] ----
    {
      const float4* src = reinterpret_cast<const float4*>(
          W + (size_t)(k0 + w_k) * N + (tile_n + w_n));
#pragma unroll
      for (int i = 0; i < 4; ++i) {
        float4 v = src[i];
        Wt[(w_n + 4 * i + 0) * LDW + w_k] = bf16_of(v.x);
        Wt[(w_n + 4 * i + 1) * LDW + w_k] = bf16_of(v.y);
        Wt[(w_n + 4 * i + 2) * LDW + w_k] = bf16_of(v.z);
        Wt[(w_n + 4 * i + 3) * LDW + w_k] = bf16_of(v.w);
      }
    }
    if (k0 + BK < K) {
      __builtin_prefetch(A + (size_t)(tile_m + a_row) * K + (k0 + BK + a_k), 0, 0);
      __builtin_prefetch(W + (size_t)(k0 + BK + w_k) * N + (tile_n + w_n), 0, 0);
    }
    __syncthreads();

    // ---- fragments ----
    BF16Frag af[2], bf[4];
#pragma unroll
    for (int mi = 0; mi < 2; ++mi) {
      const u16t* p = &As[(wm * 32 + mi * 16 + l15) * LDA];
      af[mi].h[0] = *reinterpret_cast<const u16x8*>(p + hl * 16);        // k 0..7 / 16..23
      af[mi].h[1] = *reinterpret_cast<const u16x8*>(p + 8 + hl * 16);    // k 8..15 / 24..31
    }
#pragma unroll
    for (int ni = 0; ni < 4; ++ni) {
      const u16t* p = &Wt[(wn * 64 + ni * 16 + l15) * LDW + hl * 16];
      bf[ni].h[0] = *reinterpret_cast<const u16x8*>(p);
      bf[ni].h[1] = *reinterpret_cast<const u16x8*>(p + 8);
    }
#pragma unroll
    for (int mi = 0; mi < 2; ++mi)
#pragma unroll
      for (int ni = 0; ni < 4; ++ni)
        acc[mi][ni] = wmma_bf16(af[mi].v, bf[ni].v, acc[mi][ni]);
    __syncthreads();
  }

  // ---- epilogue: bias + store ----
#pragma unroll
  for (int ni = 0; ni < 4; ++ni) {
    const int n = tile_n + wn * 64 + ni * 16 + l15;
    const float bv = bias[n];
#pragma unroll
    for (int mi = 0; mi < 2; ++mi) {
#pragma unroll
      for (int r = 0; r < 8; ++r) {
        const int m = tile_m + wm * 32 + mi * 16 + hl * 8 + r;
        const float val = acc[mi][ni][r] + bv;
        if (SPLIT_BF16) {
          const int bb = m >> 11, s = m & (SEQ - 1);
          const int h = n >> 6, d = n & (DK - 1);
          reinterpret_cast<u16t*>(outp)[((size_t)(bb * HEADS + h) * SEQ + s) * DK + d] =
              bf16_of(val);
        } else {
          reinterpret_cast<float*>(outp)[(size_t)m * N + n] = val;
        }
      }
    }
  }
}

// ---------------------------------------------------------------------------
// Flash attention, one (b,h) head slice per blockIdx.y, 64 query rows per
// block (4 waves x 16 rows). Streams 64-key blocks of K/V (bf16) through LDS:
// K via async global->LDS DMA (no VGPR round-trip), V via registers (needs
// transpose for the P*V B-fragments). Online softmax amortized over 64 keys.
// Q,K,V are bf16 [B,H,S,DK]; output f32 [B,S,D_MODEL] (heads re-concatenated).
// ---------------------------------------------------------------------------
__global__ __launch_bounds__(128)
void attn_flash(const u16t* __restrict__ Q, const u16t* __restrict__ Kb,
                const u16t* __restrict__ Vb, float* __restrict__ out) {
  constexpr int BKEY = 64;
  constexpr int LDK = DK + 8;     // 72: Ks[key][d]   (144B rows, 16B aligned)
  constexpr int LDV = BKEY + 8;   // 72: Vt[d][key]
  constexpr int LDP = BKEY + 8;   // 72: Ps[row][key]
  __shared__ alignas(16) u16t Ks[BKEY * LDK];
  __shared__ alignas(16) u16t Vt[DK * LDV];
  __shared__ alignas(16) u16t Ps[4][16 * LDP];

  const int bh = blockIdx.y;            // 0..31
  const int b = bh >> 4, h = bh & (HEADS - 1);
  const int qbase = blockIdx.x * 64;
  const int tid = threadIdx.x;
  const int lane = tid & 31, wid = tid >> 5;
  const int hl = lane >> 4, l15 = lane & 15;

  const size_t head_off = (size_t)bh * SEQ * DK;
  const u16t* Qh = Q + head_off;
  const u16t* Kh = Kb + head_off;
  const u16t* Vh = Vb + head_off;

  // Q fragments for this wave's 16 rows (A-layout), loaded once.
  BF16Frag qf[2];
  {
    const u16t* qp = Qh + (size_t)(qbase + wid * 16 + l15) * DK;
    const int ko = hl * 16;
    qf[0].h[0] = *reinterpret_cast<const u16x8*>(qp + ko);
    qf[0].h[1] = *reinterpret_cast<const u16x8*>(qp + 8 + ko);
    qf[1].h[0] = *reinterpret_cast<const u16x8*>(qp + 32 + ko);
    qf[1].h[1] = *reinterpret_cast<const u16x8*>(qp + 40 + ko);
  }

  v8f o[4] = {};
  float mrow[8], lrow[8];
#pragma unroll
  for (int r = 0; r < 8; ++r) { mrow[r] = -1e30f; lrow[r] = 0.0f; }

  const int st_key = tid >> 1;          // 0..63
  const int st_d   = (tid & 1) * 32;    // 0 or 32 (elements); 64B per thread

  for (int kb0 = 0; kb0 < SEQ; kb0 += BKEY) {
    // ---- K block: async global -> LDS (row-major, ready for QK^T B-frags) --
    {
      const u16t* gsrc = Kh + (size_t)(kb0 + st_key) * DK + st_d;
      // flat shared-pointer low 32 bits == wave-relative LDS byte address
      unsigned ldst = (unsigned)(size_t)(&Ks[st_key * LDK + st_d]);
      asm volatile(
          "global_load_async_to_lds_b128 %0, %1, off\n\t"
          "global_load_async_to_lds_b128 %0, %1, off offset:16\n\t"
          "global_load_async_to_lds_b128 %0, %1, off offset:32\n\t"
          "global_load_async_to_lds_b128 %0, %1, off offset:48"
          :: "v"(ldst), "v"(gsrc) : "memory");
    }
    // ---- V block: registers + transposed LDS stores (Vt[d][key]) ----
    {
      const u16x8* vs =
          reinterpret_cast<const u16x8*>(Vh + (size_t)(kb0 + st_key) * DK + st_d);
      u16x8 v0 = vs[0], v1 = vs[1], v2 = vs[2], v3 = vs[3];
#pragma unroll
      for (int i = 0; i < 8; ++i) Vt[(st_d + i) * LDV + st_key] = v0[i];
#pragma unroll
      for (int i = 0; i < 8; ++i) Vt[(st_d + 8 + i) * LDV + st_key] = v1[i];
#pragma unroll
      for (int i = 0; i < 8; ++i) Vt[(st_d + 16 + i) * LDV + st_key] = v2[i];
#pragma unroll
      for (int i = 0; i < 8; ++i) Vt[(st_d + 24 + i) * LDV + st_key] = v3[i];
    }
    if (kb0 + BKEY < SEQ)
      __builtin_prefetch(Vh + (size_t)(kb0 + BKEY + st_key) * DK + st_d, 0, 0);
    wait_asynccnt0();
    __syncthreads();

    // ---- S = Q * K^T : 4 key-fragments x 2 k-steps over d=64 ----
    v8f s[4] = {};
#pragma unroll
    for (int kf = 0; kf < 4; ++kf) {
#pragma unroll
      for (int kk = 0; kk < 2; ++kk) {
        BF16Frag bfr;
        const u16t* p = &Ks[(kf * 16 + l15) * LDK + kk * 32 + hl * 16];
        bfr.h[0] = *reinterpret_cast<const u16x8*>(p);
        bfr.h[1] = *reinterpret_cast<const u16x8*>(p + 8);
        s[kf] = wmma_bf16(qf[kk].v, bfr.v, s[kf]);
      }
    }
#pragma unroll
    for (int f = 0; f < 4; ++f)
#pragma unroll
      for (int r = 0; r < 8; ++r) s[f][r] *= 0.125f;   // 1/sqrt(dk)

    // ---- online softmax (rows live across 16-lane halves) ----
    float bm[8];
#pragma unroll
    for (int r = 0; r < 8; ++r)
      bm[r] = fmaxf(fmaxf(s[0][r], s[1][r]), fmaxf(s[2][r], s[3][r]));
#pragma unroll
    for (int msk = 8; msk >= 1; msk >>= 1)
#pragma unroll
      for (int r = 0; r < 8; ++r) bm[r] = fmaxf(bm[r], __shfl_xor(bm[r], msk, 32));

    float alpha[8], rs[8];
#pragma unroll
    for (int r = 0; r < 8; ++r) {
      float nm = fmaxf(mrow[r], bm[r]);
      alpha[r] = __expf(mrow[r] - nm);
      mrow[r] = nm;
      float p0 = __expf(s[0][r] - nm);
      float p1 = __expf(s[1][r] - nm);
      float p2 = __expf(s[2][r] - nm);
      float p3 = __expf(s[3][r] - nm);
      s[0][r] = p0; s[1][r] = p1; s[2][r] = p2; s[3][r] = p3;
      rs[r] = (p0 + p1) + (p2 + p3);
    }
#pragma unroll
    for (int msk = 8; msk >= 1; msk >>= 1)
#pragma unroll
      for (int r = 0; r < 8; ++r) rs[r] += __shfl_xor(rs[r], msk, 32);
#pragma unroll
    for (int r = 0; r < 8; ++r) lrow[r] = lrow[r] * alpha[r] + rs[r];
#pragma unroll
    for (int ni = 0; ni < 4; ++ni)
#pragma unroll
      for (int r = 0; r < 8; ++r) o[ni][r] *= alpha[r];

    // ---- P: C-layout f32 -> A-layout bf16 via per-wave LDS buffer ----
    {
      u16t* pw = Ps[wid];
#pragma unroll
      for (int f = 0; f < 4; ++f)
#pragma unroll
        for (int r = 0; r < 8; ++r)
          pw[(r + hl * 8) * LDP + f * 16 + l15] = bf16_of(s[f][r]);
    }
    BF16Frag pa[2];   // DS ops are in-order within a wave; waits by compiler
#pragma unroll
    for (int kc = 0; kc < 2; ++kc) {
      const u16t* p = &Ps[wid][l15 * LDP + kc * 32];
      pa[kc].h[0] = *reinterpret_cast<const u16x8*>(p + hl * 16);
      pa[kc].h[1] = *reinterpret_cast<const u16x8*>(p + 8 + hl * 16);
    }

    // ---- O += P * V : 4 d-fragments x 2 k-steps over 64 keys ----
#pragma unroll
    for (int ni = 0; ni < 4; ++ni) {
#pragma unroll
      for (int kc = 0; kc < 2; ++kc) {
        BF16Frag vf;
        const u16t* p = &Vt[(ni * 16 + l15) * LDV + kc * 32 + hl * 16];
        vf.h[0] = *reinterpret_cast<const u16x8*>(p);
        vf.h[1] = *reinterpret_cast<const u16x8*>(p + 8);
        o[ni] = wmma_bf16(pa[kc].v, vf.v, o[ni]);
      }
    }
    __syncthreads();
  }

  // ---- epilogue: normalize, write [B,S,D_MODEL] with heads concatenated ----
  const int qrow = qbase + wid * 16;
#pragma unroll
  for (int ni = 0; ni < 4; ++ni) {
    const int d = ni * 16 + l15;
#pragma unroll
    for (int r = 0; r < 8; ++r) {
      const int srow = qrow + hl * 8 + r;
      out[(size_t)(b * SEQ + srow) * D_MODEL + h * DK + d] = o[ni][r] / lrow[r];
    }
  }
}

// ---------------------------------------------------------------------------
extern "C" void kernel_launch(void* const* d_in, const int* in_sizes, int n_in,
                              void* d_out, int out_size, void* d_ws, size_t ws_size,
                              hipStream_t stream) {
  const float* query = (const float*)d_in[0];
  const float* key   = (const float*)d_in[1];
  const float* value = (const float*)d_in[2];
  const float* Wq = (const float*)d_in[3];
  const float* bq = (const float*)d_in[4];
  const float* Wk = (const float*)d_in[5];
  const float* bk = (const float*)d_in[6];
  const float* Wv = (const float*)d_in[7];
  const float* bv = (const float*)d_in[8];
  const float* Wo = (const float*)d_in[9];
  const float* bo = (const float*)d_in[10];
  float* out = (float*)d_out;

  // workspace layout (bytes): qb/kb/vb bf16 [B,H,S,DK] (8MB each), attn f32 (16MB)
  char* ws = (char*)d_ws;
  const size_t qkv_bytes = (size_t)BATCH * HEADS * SEQ * DK * sizeof(u16t);  // 8 MiB
  u16t* qb = (u16t*)(ws);
  u16t* kb = (u16t*)(ws + qkv_bytes);
  u16t* vb = (u16t*)(ws + 2 * qkv_bytes);
  float* attn = (float*)(ws + 3 * qkv_bytes);

  const dim3 gemm_grid(BATCH * SEQ / 128, D_MODEL / 128);  // 32 x 8
  const dim3 gemm_blk(256);

  gemm_bias<true><<<gemm_grid, gemm_blk, 0, stream>>>(query, Wq, bq, qb);
  gemm_bias<true><<<gemm_grid, gemm_blk, 0, stream>>>(key,   Wk, bk, kb);
  gemm_bias<true><<<gemm_grid, gemm_blk, 0, stream>>>(value, Wv, bv, vb);

  attn_flash<<<dim3(SEQ / 64, BATCH * HEADS), dim3(128), 0, stream>>>(qb, kb, vb, attn);

  gemm_bias<false><<<gemm_grid, gemm_blk, 0, stream>>>(attn, Wo, bo, out);
}
